// VQLayer_14104672600384
// MI455X (gfx1250) — compile-verified, matrix operand order checked by reading, not run on hardware
//
#include <hip/hip_runtime.h>
#include <hip/hip_bf16.h>

typedef __attribute__((ext_vector_type(16))) __bf16 v16bf;
typedef __attribute__((ext_vector_type(8)))  float  v8f;
typedef __attribute__((ext_vector_type(4)))  float  f32x4;

#define N_ROWS   262144
#define K_CODES  1024
#define D_DIM    256
#define ND       67108864ull            // N*D
#define OUT_LV   67108864ull
#define OUT_LM   67108865ull
#define OUT_IDX  67108866ull
#define STR      264                    // padded LDS row stride (bf16 elems); 528 B rows

__device__ __forceinline__ unsigned short f2bf(float x) {
    unsigned u = __float_as_uint(x);
    u += 0x7FFFu + ((u >> 16) & 1u);    // round-to-nearest-even
    return (unsigned short)(u >> 16);
}
__device__ __forceinline__ float bf2f(unsigned short h) {
    return __uint_as_float(((unsigned)h) << 16);
}

union BF16Frag {
    v16bf v;
    unsigned short u[16];
    uint4 q[2];
};

// ---------------- prep: codebook split + norms ----------------
__global__ __launch_bounds__(256) void vq_prep_cb(
    const float* __restrict__ cb, unsigned short* __restrict__ hi,
    unsigned short* __restrict__ lo, float* __restrict__ norm)
{
    int k = blockIdx.x, d = threadIdx.x;
    float x = cb[k * D_DIM + d];
    unsigned short h = f2bf(x);
    hi[k * D_DIM + d] = h;
    lo[k * D_DIM + d] = f2bf(x - bf2f(h));
    __shared__ float s[256];
    s[d] = x * x;
    __syncthreads();
    for (int t = 128; t > 0; t >>= 1) {
        if (d < t) s[d] += s[d + t];
        __syncthreads();
    }
    if (d == 0) norm[k] = s[0];
}

__global__ __launch_bounds__(256) void vq_prep_mean(
    const float* __restrict__ cb, float* __restrict__ mean_c)
{
    int d = threadIdx.x;
    float s = 0.f;
    for (int k = 0; k < K_CODES; ++k) s += cb[k * D_DIM + d];
    mean_c[d] = s * (1.0f / (float)K_CODES);
}

// Issue one chunk's codebook staging (32 codes x 256 dims, hi+lo) as async
// global->LDS B128 copies. 8 wave-instructions per wave -> ASYNCcnt += 8.
#define ISSUE_CHUNK(CH, BUF)                                                          \
    do {                                                                              \
        unsigned baseH = ldsBase + (unsigned)((BUF) * (64 * STR * 2));                \
        unsigned baseL = baseH + (unsigned)(32 * STR * 2);                            \
        _Pragma("unroll")                                                             \
        for (int i_ = 0; i_ < 4; ++i_) {                                              \
            int g_ = i_ * 256 + tid;                                                  \
            int code_ = g_ >> 5, col_ = g_ & 31;                                      \
            unsigned ldsOff = (unsigned)(code_ * (STR * 2) + col_ * 16);              \
            unsigned long long gb =                                                   \
                ((unsigned long long)(((CH) * 32 + code_) * 32 + col_)) * 16ull;      \
            unsigned long long gaH = (unsigned long long)(uintptr_t)cb_hi + gb;       \
            unsigned long long gaL = (unsigned long long)(uintptr_t)cb_lo + gb;       \
            unsigned dH = baseH + ldsOff;                                             \
            unsigned dL = baseL + ldsOff;                                             \
            asm volatile("global_load_async_to_lds_b128 %0, %1, off"                  \
                         :: "v"(dH), "v"(gaH) : "memory");                            \
            asm volatile("global_load_async_to_lds_b128 %0, %1, off"                  \
                         :: "v"(dL), "v"(gaL) : "memory");                            \
        }                                                                             \
    } while (0)

// ---------------- main argmin kernel (bf16x3 WMMA, async double-buffered) ----------
__global__ __launch_bounds__(256) void vq_argmin_kernel(
    const float* __restrict__ z_e,
    const unsigned short* __restrict__ cb_hi,
    const unsigned short* __restrict__ cb_lo,
    const float* __restrict__ cb_norm,
    int* __restrict__ best_idx)
{
    extern __shared__ unsigned short dynsmem[];   // 2 buffers x (hi 32*STR + lo 32*STR)
    __shared__ float sCbn[K_CODES];

    const int tid  = threadIdx.x;
    const int wave = tid >> 5;
    const int lane = tid & 31;
    const int l15  = lane & 15;
    const int half = lane >> 4;
    const int rowBase = blockIdx.x * 128 + wave * 16;
    const int myRow   = rowBase + l15;
    const unsigned ldsBase = (unsigned)(uintptr_t)(void*)dynsmem;  // LDS byte offset

    for (int i = tid; i < K_CODES; i += 256) sCbn[i] = cb_norm[i];

    // Kick off chunk 0 staging immediately; it lands while we convert A-frags.
    ISSUE_CHUNK(0, 0);

    // Load this wave's 16 rows of z_e (full D) once; split into bf16 hi/lo frags.
    // A frag layout (16-bit, 16x32): lane<16 -> K in {0..7}u{16..23}+k0, lane>=16 -> +8.
    BF16Frag ah[8], al[8];
    const f32x4* zr = (const f32x4*)(z_e + (size_t)myRow * D_DIM);
    #pragma unroll
    for (int kc = 0; kc < 8; ++kc) {
        const int k0 = kc * 32 + half * 8;
        f32x4 f0 = __builtin_nontemporal_load(&zr[k0 / 4]);
        f32x4 f1 = __builtin_nontemporal_load(&zr[k0 / 4 + 1]);
        f32x4 f2 = __builtin_nontemporal_load(&zr[k0 / 4 + 4]);
        f32x4 f3 = __builtin_nontemporal_load(&zr[k0 / 4 + 5]);
        float a_[16] = { f0.x, f0.y, f0.z, f0.w, f1.x, f1.y, f1.z, f1.w,
                         f2.x, f2.y, f2.z, f2.w, f3.x, f3.y, f3.z, f3.w };
        #pragma unroll
        for (int j = 0; j < 16; ++j) {
            unsigned short h = f2bf(a_[j]);
            ah[kc].u[j] = h;
            al[kc].u[j] = f2bf(a_[j] - bf2f(h));
        }
    }

    float bd[8];
    int   bix[8];
    #pragma unroll
    for (int r = 0; r < 8; ++r) { bd[r] = 3.4e38f; bix[r] = 0; }

    asm volatile("s_wait_asynccnt 0x0" ::: "memory");
    __syncthreads();   // chunk 0 fully staged by all waves

    for (int chunk = 0; chunk < 32; ++chunk) {
        const int cur = chunk & 1;
        if (chunk + 1 < 32) ISSUE_CHUNK(chunk + 1, cur ^ 1);   // prefetch next chunk

        const unsigned short* sBh = dynsmem + (size_t)cur * (64 * STR);
        const unsigned short* sBl = sBh + 32 * STR;

        v8f acc0 = {}, acc1 = {};
        #pragma unroll
        for (int kc = 0; kc < 8; ++kc) {
            const int kd = kc * 32 + half * 8;
            BF16Frag bh, bl;
            bh.q[0] = *(const uint4*)&sBh[l15 * STR + kd];
            bh.q[1] = *(const uint4*)&sBh[l15 * STR + kd + 16];
            bl.q[0] = *(const uint4*)&sBl[l15 * STR + kd];
            bl.q[1] = *(const uint4*)&sBl[l15 * STR + kd + 16];
            acc0 = __builtin_amdgcn_wmma_f32_16x16x32_bf16(false, al[kc].v, false, bh.v, (short)0, acc0, false, false);
            acc0 = __builtin_amdgcn_wmma_f32_16x16x32_bf16(false, ah[kc].v, false, bl.v, (short)0, acc0, false, false);
            acc0 = __builtin_amdgcn_wmma_f32_16x16x32_bf16(false, ah[kc].v, false, bh.v, (short)0, acc0, false, false);
            const int c1 = 16 + l15;
            bh.q[0] = *(const uint4*)&sBh[c1 * STR + kd];
            bh.q[1] = *(const uint4*)&sBh[c1 * STR + kd + 16];
            bl.q[0] = *(const uint4*)&sBl[c1 * STR + kd];
            bl.q[1] = *(const uint4*)&sBl[c1 * STR + kd + 16];
            acc1 = __builtin_amdgcn_wmma_f32_16x16x32_bf16(false, al[kc].v, false, bh.v, (short)0, acc1, false, false);
            acc1 = __builtin_amdgcn_wmma_f32_16x16x32_bf16(false, ah[kc].v, false, bl.v, (short)0, acc1, false, false);
            acc1 = __builtin_amdgcn_wmma_f32_16x16x32_bf16(false, ah[kc].v, false, bh.v, (short)0, acc1, false, false);
        }

        // dist = ||c||^2 - 2*dot ; C layout: VGPR r -> row r (lanes<16) / 8+r; col = l15
        #pragma unroll
        for (int r = 0; r < 8; ++r) {
            int code0 = chunk * 32 + l15;
            float d0 = sCbn[code0] - 2.0f * acc0[r];
            if (d0 < bd[r]) { bd[r] = d0; bix[r] = code0; }
            int code1 = chunk * 32 + 16 + l15;
            float d1 = sCbn[code1] - 2.0f * acc1[r];
            if (d1 < bd[r]) { bd[r] = d1; bix[r] = code1; }
        }

        // My prefetch for the next buffer has landed; barrier makes it visible to all
        // waves and guarantees everyone is done reading buf[cur] before it is reused.
        asm volatile("s_wait_asynccnt 0x0" ::: "memory");
        __syncthreads();
    }

    // Reduce across the 16 columns (lanes sharing the same half).
    #pragma unroll
    for (int r = 0; r < 8; ++r) {
        float d = bd[r];
        int   bi = bix[r];
        #pragma unroll
        for (int m = 1; m <= 8; m <<= 1) {
            float od = __shfl_xor(d, m, 32);
            int   oi = __shfl_xor(bi, m, 32);
            if (od < d || (od == d && oi < bi)) { d = od; bi = oi; }
        }
        if (l15 == 0) best_idx[rowBase + half * 8 + r] = bi;
    }
}

// ---------------- gather z_q, indices, loss partials ----------------
__global__ __launch_bounds__(256) void vq_gather_kernel(
    const float* __restrict__ z_e, const float* __restrict__ cb,
    const int* __restrict__ best_idx, const float* __restrict__ mean_c,
    float* __restrict__ out, float2* __restrict__ partials)
{
    const int tid = threadIdx.x;
    const float mc = mean_c[tid];
    float lv = 0.f, lm = 0.f;
    const int r0 = blockIdx.x * 8;
    #pragma unroll
    for (int i = 0; i < 8; ++i) {
        int row = r0 + i;
        int idx = best_idx[row];
        float z = __builtin_nontemporal_load(&z_e[(size_t)row * D_DIM + tid]);
        float c = cb[(size_t)idx * D_DIM + tid];
        __builtin_nontemporal_store(c, &out[(size_t)row * D_DIM + tid]); // z_q_st fwd
        float dv = c - z;  lv += dv * dv;
        float dm = z - mc; lm += dm * dm;
        if (tid == 0) out[OUT_IDX + row] = (float)idx;
    }
    __shared__ float s1[256], s2[256];
    s1[tid] = lv; s2[tid] = lm;
    __syncthreads();
    for (int t = 128; t > 0; t >>= 1) {
        if (tid < t) { s1[tid] += s1[tid + t]; s2[tid] += s2[tid + t]; }
        __syncthreads();
    }
    if (tid == 0) partials[blockIdx.x] = make_float2(s1[0], s2[0]);
}

__global__ __launch_bounds__(256) void vq_finalize_kernel(
    const float2* __restrict__ partials, float* __restrict__ out)
{
    const int tid = threadIdx.x;
    float lv = 0.f, lm = 0.f;
    for (int i = tid; i < 32768; i += 256) {
        float2 p = partials[i];
        lv += p.x; lm += p.y;
    }
    __shared__ float s1[256], s2[256];
    s1[tid] = lv; s2[tid] = lm;
    __syncthreads();
    for (int t = 128; t > 0; t >>= 1) {
        if (tid < t) { s1[tid] += s1[tid + t]; s2[tid] += s2[tid + t]; }
        __syncthreads();
    }
    if (tid == 0) {
        const float inv = 1.0f / (float)ND;
        out[OUT_LV] = s1[0] * inv;
        out[OUT_LM] = s2[0] * inv;
    }
}

extern "C" void kernel_launch(void* const* d_in, const int* in_sizes, int n_in,
                              void* d_out, int out_size, void* d_ws, size_t ws_size,
                              hipStream_t stream) {
    const float* z_e = (const float*)d_in[0];
    const float* cb  = (const float*)d_in[1];
    float* out = (float*)d_out;

    char* w = (char*)d_ws;
    unsigned short* cb_hi = (unsigned short*)(w);             // 512 KB
    unsigned short* cb_lo = (unsigned short*)(w + 524288);    // 512 KB
    float* cb_norm        = (float*)(w + 1048576);            // 4 KB
    float* mean_c         = (float*)(w + 1052672);            // 1 KB
    int*   best           = (int*)(w + 1053696);              // 1 MB
    float2* partials      = (float2*)(w + 2102272);           // 256 KB

    const size_t dynLds = 2ull * 64 * STR * sizeof(unsigned short);  // 67584 B

    vq_prep_cb<<<K_CODES, 256, 0, stream>>>(cb, cb_hi, cb_lo, cb_norm);
    vq_prep_mean<<<1, 256, 0, stream>>>(cb, mean_c);
    vq_argmin_kernel<<<N_ROWS / 128, 256, dynLds, stream>>>(z_e, cb_hi, cb_lo, cb_norm, best);
    vq_gather_kernel<<<N_ROWS / 8, 256, 0, stream>>>(z_e, cb, best, mean_c, out, partials);
    vq_finalize_kernel<<<1, 256, 0, stream>>>(partials, out);
}
